// SocialReverberationLayer_15126874817180
// MI455X (gfx1250) — compile-verified, matrix operand order checked by reading, not run on hardware
//
#include <hip/hip_runtime.h>

// ---------------------------------------------------------------------------
// Problem constants (from reference)
// ---------------------------------------------------------------------------
#define B_     64
#define K_     10
#define OBS_   8
#define PRED_  12
#define P_     8
#define STEPS_ 64
#define D_     128
#define H_     8
#define DFF_   512
#define DN_    16
#define KC_    20
#define TCH_   2
#define KB_    (K_ * B_)

#define EMB_SCALE 11.313708499f   // sqrt(128)

typedef _Float16 h16;
typedef __attribute__((ext_vector_type(16))) _Float16 v16h;
typedef __attribute__((ext_vector_type(8)))  _Float16 v8h;
typedef __attribute__((ext_vector_type(8)))  float    v8f;
typedef __attribute__((ext_vector_type(4)))  unsigned int v4u;
typedef __attribute__((ext_vector_type(8)))  int      v8i;
typedef __attribute__((ext_vector_type(4)))  int      v4i;

// Tensor Data Mover path (probe-verified builtin; arity differs per toolchain)
#if defined(__gfx1250__) && defined(__has_builtin)
#if __has_builtin(__builtin_amdgcn_tensor_load_to_lds) && \
    __has_builtin(__builtin_amdgcn_s_wait_tensorcnt)
#define USE_TDM 1
#endif
#endif

// LDS layout (halves):  sE 8192 | sD 8192 | sQ 8192 | sK 8192 | sV 8192 | sS 32768
#define SMEM_HALVES (8192 * 5 + 32768)
#define SMEM_BYTES  (SMEM_HALVES * 2)

// ---------------------------------------------------------------------------
// Parameter structs (device pointers; f16 weights pre-transposed to N x K)
// ---------------------------------------------------------------------------
struct AttnWD { const h16 *wq, *wk, *wv, *wo; const float *bq, *bk, *bv, *bo; };
struct FfnWD  { const h16 *w1, *w2; const float *b1, *b2; };
struct EncWD  { AttnWD attn; FfnWD ffn; const float *ln0g, *ln0b, *ln1g, *ln1b; };
struct DecWD  { AttnWD attn, cross; FfnWD ffn;
                const float *ln0g, *ln0b, *ln1g, *ln1b, *ln2g, *ln2b; };
struct TArgs {
  const h16*   x_in;      // (KB, 64, 128) f16
  const float* x_ego;     // (B, 8, 2)
  const float* pe;        // (64, 128) precomputed positional encoding
  const h16*   emb_in_t;  const float* bemb_in;
  const float* emb_tg;    const float* bemb_tg;   // (2,128) fp32, tiny
  EncWD enc[2];
  DecWD dec[2];
  const float *wk1, *bk1, *wk2, *bk2, *wdec, *bdec;
  float* out;             // (B, K*KC, PRED, TCH)
};

// ---------------------------------------------------------------------------
// WMMA helpers: f32 = f16(16x32) x f16(32x16) + f32  (wave32)
// A layout (ISA 7.12.2): lane<16 -> elems 0..7 = K0..7, 8..15 = K16..23
//                        lane>=16 -> elems 0..7 = K8..15, 8..15 = K24..31
// ---------------------------------------------------------------------------
__device__ __forceinline__ v8f wmma_f16(v16h a, v16h b, v8f c) {
  return __builtin_amdgcn_wmma_f32_16x16x32_f16(false, a, false, b,
                                                (short)0, c, false, false);
}

// A-fragment (and B-fragment from pre-transposed NxK weights): two b128 loads
__device__ __forceinline__ v16h frag_ld(const h16* p, int row0, int k0, int ld) {
  const int lane = threadIdx.x & 31;
  const int m = lane & 15, kh = (lane >> 4) << 3;
  const h16* r = p + (size_t)(row0 + m) * ld + k0 + kh;
  v8h lo = *(const v8h*)r;
  v8h hi = *(const v8h*)(r + 16);
  v16h a;
#pragma unroll
  for (int i = 0; i < 8; ++i) { a[i] = lo[i]; a[8 + i] = hi[i]; }
  return a;
}

// Fragment with real contraction length 16 (head dim), upper half zero
__device__ __forceinline__ v16h frag_ld_k16(const h16* p, int row0, int c0, int ld) {
  const int lane = threadIdx.x & 31;
  const int m = lane & 15, kh = (lane >> 4) << 3;
  v8h lo = *(const v8h*)(p + (size_t)(row0 + m) * ld + c0 + kh);
  v16h a;
#pragma unroll
  for (int i = 0; i < 8; ++i) { a[i] = lo[i]; a[8 + i] = (h16)0.f; }
  return a;
}

// B-fragment from a row-major (keys x dims) V tile: contraction = key index
__device__ __forceinline__ v16h frag_ld_bT(const h16* vb, int k0, int c0, int ld) {
  const int lane = threadIdx.x & 31;
  const int nn = lane & 15, kh = (lane >> 4) << 3;
  v16h b;
#pragma unroll
  for (int i = 0; i < 8; ++i) {
    b[i]     = vb[(size_t)(k0 + kh + i) * ld + c0 + nn];
    b[8 + i] = vb[(size_t)(k0 + kh + 16 + i) * ld + c0 + nn];
  }
  return b;
}

// ---------------------------------------------------------------------------
// 64xN = (64xK f16 LDS) x (Wt: NxK f16 global), per-wave column strips
// ---------------------------------------------------------------------------
template <int N, int K, bool RELU>
__device__ void gemm64_store(const h16* __restrict__ X, const h16* __restrict__ Wt,
                             const float* __restrict__ bias, h16* __restrict__ OUT) {
  const int wv = threadIdx.x >> 5, lane = threadIdx.x & 31;
  for (int nt = wv; nt < N / 16; nt += 8) {
    for (int mt = 0; mt < 4; ++mt) {
      v8f acc = {};
#pragma unroll
      for (int kk = 0; kk < K; kk += 32) {
        v16h a = frag_ld(X, mt * 16, kk, K);
        v16h b = frag_ld(Wt, nt * 16, kk, K);
        acc = wmma_f16(a, b, acc);
      }
      const int col = nt * 16 + (lane & 15), mh = (lane >> 4) << 3;
      const float bv = bias[col];
#pragma unroll
      for (int vv = 0; vv < 8; ++vv) {
        float r = acc[vv] + bv;
        if (RELU) r = fmaxf(r, 0.f);
        OUT[(size_t)(mt * 16 + mh + vv) * N + col] = (h16)r;
      }
    }
  }
}

// out += residual: STATE(64x128) = STATE + X @ Wt + bias  (pre-LN value)
template <int K>
__device__ void gemm64_residual(const h16* __restrict__ X, const h16* __restrict__ Wt,
                                const float* __restrict__ bias, h16* __restrict__ STATE) {
  const int wv = threadIdx.x >> 5, lane = threadIdx.x & 31;
  const int nt = wv;
  for (int mt = 0; mt < 4; ++mt) {
    v8f acc = {};
#pragma unroll
    for (int kk = 0; kk < K; kk += 32) {
      v16h a = frag_ld(X, mt * 16, kk, K);
      v16h b = frag_ld(Wt, nt * 16, kk, K);
      acc = wmma_f16(a, b, acc);
    }
    const int col = nt * 16 + (lane & 15), mh = (lane >> 4) << 3;
    const float bv = bias[col];
#pragma unroll
    for (int vv = 0; vv < 8; ++vv) {
      const int idx = (mt * 16 + mh + vv) * D_ + col;
      STATE[idx] = (h16)((float)STATE[idx] + acc[vv] + bv);
    }
  }
}

// embedding: OUT = (X @ Wt + b) * sqrt(D) + PE  (PE from precomputed table)
__device__ void gemm_embed(const h16* X, const h16* Wt, const float* bias,
                           const float* pe, h16* OUT) {
  const int wv = threadIdx.x >> 5, lane = threadIdx.x & 31;
  const int nt = wv;
  for (int mt = 0; mt < 4; ++mt) {
    v8f acc = {};
#pragma unroll
    for (int kk = 0; kk < D_; kk += 32) {
      v16h a = frag_ld(X, mt * 16, kk, D_);
      v16h b = frag_ld(Wt, nt * 16, kk, D_);
      acc = wmma_f16(a, b, acc);
    }
    const int col = nt * 16 + (lane & 15), mh = (lane >> 4) << 3;
    const float bv = bias[col];
#pragma unroll
    for (int vv = 0; vv < 8; ++vv) {
      const int row = mt * 16 + mh + vv;
      OUT[row * D_ + col] = (h16)((acc[vv] + bv) * EMB_SCALE + pe[row * D_ + col]);
    }
  }
}

// LayerNorm over rows of 64x128 f16 state, wave32 shuffle reduction
__device__ void layernorm(h16* S, const float* g, const float* bb) {
  const int wv = threadIdx.x >> 5, lane = threadIdx.x & 31;
  for (int r = wv; r < 64; r += 8) {
    float x[4];
#pragma unroll
    for (int j = 0; j < 4; ++j) x[j] = (float)S[r * D_ + lane + 32 * j];
    float s = x[0] + x[1] + x[2] + x[3];
#pragma unroll
    for (int o = 16; o > 0; o >>= 1) s += __shfl_xor(s, o, 32);
    const float mu = s * (1.f / 128.f);
    float v2 = 0.f;
#pragma unroll
    for (int j = 0; j < 4; ++j) { x[j] -= mu; v2 += x[j] * x[j]; }
#pragma unroll
    for (int o = 16; o > 0; o >>= 1) v2 += __shfl_xor(v2, o, 32);
    const float rs = rsqrtf(v2 * (1.f / 128.f) + 1e-6f);
#pragma unroll
    for (int j = 0; j < 4; ++j) {
      const int c = lane + 32 * j;
      S[r * D_ + c] = (h16)(x[j] * rs * g[c] + bb[c]);
    }
  }
}

// ---------------------------------------------------------------------------
// Multi-head attention; 8 waves == 8 heads, attention itself is wave-local.
// STATE (residual target) updated in place with pre-LN value.
// ---------------------------------------------------------------------------
__device__ void mha_layer(const h16* Xq, const h16* Xkv,
                          h16* sQ, h16* sK, h16* sV, h16* sS,
                          const AttnWD& w, h16* STATE, bool causal) {
  gemm64_store<D_, D_, false>(Xq,  w.wq, w.bq, sQ);  // wave h -> head-h cols
  gemm64_store<D_, D_, false>(Xkv, w.wk, w.bk, sK);
  gemm64_store<D_, D_, false>(Xkv, w.wv, w.bv, sV);

  const int h = threadIdx.x >> 5, lane = threadIdx.x & 31;
  h16* S = sS + h * 64 * 64;

  // scores: S_h = Q_h (64x16) @ K_h^T, contraction padded 16 -> 32
  for (int mt = 0; mt < 4; ++mt) {
    v16h a = frag_ld_k16(sQ, mt * 16, h * 16, D_);
    for (int nt = 0; nt < 4; ++nt) {
      v16h b = frag_ld_k16(sK, nt * 16, h * 16, D_);
      v8f acc = {};
      acc = wmma_f16(a, b, acc);
      const int col = nt * 16 + (lane & 15), mh = (lane >> 4) << 3;
#pragma unroll
      for (int vv = 0; vv < 8; ++vv) S[(mt * 16 + mh + vv) * 64 + col] = (h16)acc[vv];
    }
  }

  // softmax over rows (scale 1/sqrt(16), optional causal mask), wave-local
  for (int r = 0; r < 64; ++r) {
    float v0 = 0.25f * (float)S[r * 64 + lane];
    float v1 = 0.25f * (float)S[r * 64 + lane + 32];
    if (causal) { if (lane > r) v0 = -1e9f; if (lane + 32 > r) v1 = -1e9f; }
    float m = fmaxf(v0, v1);
#pragma unroll
    for (int o = 16; o > 0; o >>= 1) m = fmaxf(m, __shfl_xor(m, o, 32));
    const float e0 = __expf(v0 - m), e1 = __expf(v1 - m);
    float sm2 = e0 + e1;
#pragma unroll
    for (int o = 16; o > 0; o >>= 1) sm2 += __shfl_xor(sm2, o, 32);
    const float inv = 1.f / sm2;
    S[r * 64 + lane]      = (h16)(e0 * inv);
    S[r * 64 + lane + 32] = (h16)(e1 * inv);
  }

  // O_h = P (64x64) @ V_h (64x16), 2 k-steps; write into sQ head columns
  for (int mt = 0; mt < 4; ++mt) {
    v8f acc = {};
#pragma unroll
    for (int k0 = 0; k0 < 64; k0 += 32) {
      v16h a = frag_ld(S, mt * 16, k0, 64);
      v16h b = frag_ld_bT(sV, k0, h * 16, D_);
      acc = wmma_f16(a, b, acc);
    }
    const int col = h * 16 + (lane & 15), mh = (lane >> 4) << 3;
#pragma unroll
    for (int vv = 0; vv < 8; ++vv) sQ[(mt * 16 + mh + vv) * D_ + col] = (h16)acc[vv];
  }
  __syncthreads();                       // all heads' O ready; Xq/Xkv reads done
  gemm64_residual<D_>(sQ, w.wo, w.bo, STATE);
  __syncthreads();
}

__device__ void enc_layer(h16* sE, h16* sQ, h16* sK, h16* sV, h16* sS, const EncWD& L) {
  mha_layer(sE, sE, sQ, sK, sV, sS, L.attn, sE, false);
  layernorm(sE, L.ln0g, L.ln0b); __syncthreads();
  gemm64_store<DFF_, D_, true>(sE, L.ffn.w1, L.ffn.b1, sS);   // sS as 64x512
  __syncthreads();
  gemm64_residual<DFF_>(sS, L.ffn.w2, L.ffn.b2, sE);
  __syncthreads();
  layernorm(sE, L.ln1g, L.ln1b); __syncthreads();
}

__device__ void dec_layer(h16* sD, const h16* sE, h16* sQ, h16* sK, h16* sV, h16* sS,
                          const DecWD& L) {
  mha_layer(sD, sD, sQ, sK, sV, sS, L.attn, sD, true);        // causal self-attn
  layernorm(sD, L.ln0g, L.ln0b); __syncthreads();
  mha_layer(sD, sE, sQ, sK, sV, sS, L.cross, sD, false);      // cross-attn
  layernorm(sD, L.ln1g, L.ln1b); __syncthreads();
  gemm64_store<DFF_, D_, true>(sD, L.ffn.w1, L.ffn.b1, sS);
  __syncthreads();
  gemm64_residual<DFF_>(sS, L.ffn.w2, L.ffn.b2, sD);
  __syncthreads();
  layernorm(sD, L.ln2g, L.ln2b); __syncthreads();
}

// ---------------------------------------------------------------------------
// One workgroup (256 threads = 8 waves) per sequence. Full transformer +
// head contractions fused; f_tran never leaves LDS.
// ---------------------------------------------------------------------------
__global__ void __launch_bounds__(256)
k_transformer(TArgs A) {
  extern __shared__ h16 sm[];
  h16* sE = sm;
  h16* sD = sm + 8192;
  h16* sQ = sm + 16384;
  h16* sK = sm + 24576;
  h16* sV = sm + 32768;
  h16* sS = sm + 40960;                 // 32768 halves (scores / FFN hidden)

  const int n = blockIdx.x;             // sequence = k*B + b
  const int b = n % B_, kidx = n / B_;

  // warm WGP$/L2 with first encoder weights (global_prefetch_b8)
  __builtin_prefetch(A.enc[0].attn.wq + (size_t)threadIdx.x * 64, 0, 1);

  // stage x_in (contiguous 16KB tile) into LDS
  const h16* xin = A.x_in + (size_t)n * STEPS_ * D_;
#if defined(USE_TDM)
  // Tensor Data Mover: 1-row 2D descriptor, data_size=2B, tile_dim0=8192
  if (threadIdx.x < 32) {
    const unsigned long long ga = (unsigned long long)(size_t)xin;
    const unsigned int ldsa = (unsigned int)(size_t)sQ;   // LDS aperture offset
    v4u g0;
    g0[0] = 1u;                                          // count=1 (valid D#)
    g0[1] = ldsa;                                        // lds_addr [63:32]
    g0[2] = (unsigned int)ga;                            // global_addr [95:64]
    g0[3] = (unsigned int)((ga >> 32) & 0x1FFFFFFu) | (2u << 30); // addr hi + type=2
    v8i g1 = {};
    g1[0] = 0x00010000;        // wg_mask=0, data_size=1 (2 bytes)
    g1[1] = 0x20000000;        // tensor_dim0 = 8192 (bits 79:48, low half)
    g1[2] = 0x00010000;        // tensor_dim1 = 1 (bits 111:80, low half)
    g1[3] = 0x20000000;        // tile_dim0 = 8192 (bits 127:112)
    g1[4] = 1;                 // tile_dim1 = 1
    g1[5] = 8192;              // tensor_dim0_stride = 8192 elements
    v4i g2 = {}, g3 = {};
#if __clang_major__ >= 23
    v8i g4 = {};
    __builtin_amdgcn_tensor_load_to_lds(g0, g1, g2, g3, g4, 0);
#else
    __builtin_amdgcn_tensor_load_to_lds(g0, g1, g2, g3, 0);
#endif
    __builtin_amdgcn_s_wait_tensorcnt(0);
  }
  __syncthreads();
#else
  for (int i = threadIdx.x; i < 8192; i += 256) sQ[i] = xin[i];
  __syncthreads();
#endif

  gemm_embed(sQ, A.emb_in_t, A.bemb_in, A.pe, sE);
  __syncthreads();

  enc_layer(sE, sQ, sK, sV, sS, A.enc[0]);
  enc_layer(sE, sQ, sK, sV, sS, A.enc[1]);

  // decoder target embedding: x_tg[s] = x_ego_diff[b, s/8, :] (TCH=2) -> tiny VALU
  for (int i = threadIdx.x; i < 8192; i += 256) {
    const int s = i >> 7, c = i & 127;
    const float* xe = A.x_ego + ((size_t)b * OBS_ + (s >> 3)) * 2;
    const float vv = (xe[0] * A.emb_tg[c] + xe[1] * A.emb_tg[128 + c] +
                      A.bemb_tg[c]) * EMB_SCALE + A.pe[i];
    sD[i] = (h16)vv;
  }
  __syncthreads();

  dec_layer(sD, sE, sQ, sK, sV, sS, A.dec[0]);
  dec_layer(sD, sE, sQ, sK, sV, sS, A.dec[1]);

  // ---------------- head: k1/k2, s = f^T k1, u = f^T k2, y = (s x u) @ wdec
  float* k1f = (float*)sS;              // 64 x 20
  float* k2f = k1f + 64 * KC_;          // 64 x 12
  float* sF  = k2f + 64 * PRED_;        // 128 x 20
  float* uF  = sF + D_ * KC_;           // 128 x 12

  for (int i = threadIdx.x; i < 64 * 32; i += 256) {
    const int r = i >> 5, c = i & 31;
    if (c < KC_) {
      float a = A.bk1[c];
      for (int j = 0; j < D_; ++j) a += (float)sD[r * D_ + j] * A.wk1[j * KC_ + c];
      k1f[r * KC_ + c] = tanhf(a);
    } else if (c - KC_ < PRED_) {
      const int t = c - KC_;
      float a = A.bk2[t];
      for (int j = 0; j < D_; ++j) a += (float)sD[r * D_ + j] * A.wk2[j * PRED_ + t];
      k2f[r * PRED_ + t] = tanhf(a);
    }
  }
  __syncthreads();

  for (int i = threadIdx.x; i < D_ * 32; i += 256) {
    const int d = i >> 5, c = i & 31;
    if (c < KC_) {
      float a = 0.f;
      for (int r = 0; r < 64; ++r) a += (float)sD[r * D_ + d] * k1f[r * KC_ + c];
      sF[d * KC_ + c] = a;
    } else if (c - KC_ < PRED_) {
      const int t = c - KC_;
      float a = 0.f;
      for (int r = 0; r < 64; ++r) a += (float)sD[r * D_ + d] * k2f[r * PRED_ + t];
      uF[d * PRED_ + t] = a;
    }
  }
  __syncthreads();

  for (int i = threadIdx.x; i < KC_ * PRED_ * TCH_; i += 256) {
    const int o = i % 2, t = (i / 2) % PRED_, c = i / (2 * PRED_);
    float a = A.bdec[o];
    for (int d = 0; d < D_; ++d) a += sF[d * KC_ + c] * uF[d * PRED_ + t] * A.wdec[d * 2 + o];
    A.out[(((size_t)b * (K_ * KC_) + kidx * KC_ + c) * PRED_ + t) * TCH_ + o] = a;
  }
}

// ---------------------------------------------------------------------------
// Prep kernels
// ---------------------------------------------------------------------------
// positional encoding table (64 x 128), computed once
__global__ void k_pe(float* __restrict__ pe) {
  const int i = blockIdx.x * 256 + threadIdx.x;
  if (i >= STEPS_ * D_) return;
  const int s = i >> 7, c = i & 127;
  const float expo  = (float)(2 * (c >> 1)) * (1.0f / 128.0f);
  const float angle = (float)s / __powf(10000.0f, expo);
  pe[i] = ((c & 1) == 0) ? __sinf(angle) : __cosf(angle);
}

// fp32 (K x N) -> f16 transposed (N x K) for b128 B-fragment loads
__global__ void k_convert_wt(const float* __restrict__ W, h16* __restrict__ Wt,
                             int K, int N) {
  const int i = blockIdx.x * 256 + threadIdx.x;
  if (i >= K * N) return;
  const int nn = i / K, kk = i % K;
  Wt[(size_t)nn * K + kk] = (h16)W[(size_t)kk * N + nn];
}

// f_behavior[b,s,c] = relu( [f_ego_diff[b,s/8]; re_matrix[b,s/8,s%8]] @ wc + bc )
__global__ void k_fbehavior(const float* __restrict__ f_ego, const float* __restrict__ re,
                            const float* __restrict__ wc, const float* __restrict__ bc,
                            h16* __restrict__ fb) {
  const int i = blockIdx.x * 256 + threadIdx.x;
  if (i >= B_ * STEPS_ * 64) return;
  const int c = i & 63, s = (i >> 6) & 63, b = i >> 12;
  const float* fe = f_ego + ((size_t)b * OBS_ + (s >> 3)) * 64;
  const float* rr = re + (((size_t)b * OBS_ + (s >> 3)) * P_ + (s & 7)) * 64;
  float acc = bc[c];
  for (int j = 0; j < 64; ++j) acc += fe[j] * wc[j * 64 + c];
  for (int j = 0; j < 64; ++j) acc += rr[j] * wc[(64 + j) * 64 + c];
  fb[i] = (h16)fmaxf(acc, 0.f);
}

// x_in[n,s,0:64] = f_behavior[b,s,:], x_in[n,s,64:128] = tanh(z @ wie + bie)
__global__ void k_xin(const h16* __restrict__ fb, const float* __restrict__ z,
                      const float* __restrict__ wie, const float* __restrict__ bie,
                      h16* __restrict__ x_in) {
  const int i = blockIdx.x * 256 + threadIdx.x;
  if (i >= KB_ * STEPS_ * 64) return;
  const int c = i & 63, s = (i >> 6) & 63, n = i >> 12;
  const int b = n % B_;
  const float* zz = z + ((size_t)n * STEPS_ + s) * DN_;
  float acc = bie[c];
  for (int j = 0; j < DN_; ++j) acc += zz[j] * wie[j * 64 + c];
  h16* row = x_in + ((size_t)n * STEPS_ + s) * D_;
  row[c]      = fb[((size_t)b * STEPS_ + s) * 64 + c];
  row[64 + c] = (h16)tanhf(acc);
}

// ---------------------------------------------------------------------------
// Host launcher. Param leaves follow setup_inputs() insertion order:
//   0:x_ego_diff 1:f_ego_diff 2:re_matrix 3:z then params leaves:
//   wc,bc,wie,bie,emb_in,bemb_in,emb_tg,bemb_tg,
//   enc[0..1]{wq,wk,wv,wo,bq,bk,bv,bo,w1,b1,w2,b2,ln0_g,ln0_b,ln1_g,ln1_b},
//   dec[0..1]{attn(8),w1,b1,w2,b2,cross(8),ln0_g..ln2_b},
//   wk1,bk1,wk2,bk2,wdec,bdec
// ---------------------------------------------------------------------------
extern "C" void kernel_launch(void* const* d_in, const int* in_sizes, int n_in,
                              void* d_out, int out_size, void* d_ws, size_t ws_size,
                              hipStream_t stream) {
  (void)in_sizes; (void)n_in; (void)out_size; (void)ws_size;

  const float* x_ego = (const float*)d_in[0];
  const float* f_ego = (const float*)d_in[1];
  const float* re    = (const float*)d_in[2];
  const float* z     = (const float*)d_in[3];

  int ci = 4;
  auto nf = [&]() -> const float* { return (const float*)d_in[ci++]; };

  const float* wc      = nf();  const float* bc      = nf();
  const float* wie     = nf();  const float* bie     = nf();
  const float* emb_in  = nf();  const float* bemb_in = nf();
  const float* emb_tg  = nf();  const float* bemb_tg = nf();

  // workspace carve-out (halves)
  h16* wsH = (h16*)d_ws;
  size_t off = 0;
  auto wsalloc = [&](size_t nh) { h16* p = wsH + off; off += nh; return p; };
  h16*   x_in = wsalloc((size_t)KB_ * STEPS_ * D_);
  h16*   fb   = wsalloc((size_t)B_ * STEPS_ * 64);
  float* pe   = (float*)wsalloc((size_t)STEPS_ * D_ * 2);   // 8192 floats

  auto cvt = [&](const float* W, int Kd, int Nd) -> const h16* {
    h16* dst = wsalloc((size_t)Kd * Nd);
    const int tot = Kd * Nd;
    k_convert_wt<<<(tot + 255) / 256, 256, 0, stream>>>(W, dst, Kd, Nd);
    return dst;
  };
  auto cvt_attn = [&](AttnWD& a) {
    a.wq = cvt(nf(), D_, D_); a.wk = cvt(nf(), D_, D_);
    a.wv = cvt(nf(), D_, D_); a.wo = cvt(nf(), D_, D_);
    a.bq = nf(); a.bk = nf(); a.bv = nf(); a.bo = nf();
  };

  TArgs A{};
  A.x_in = x_in;  A.x_ego = x_ego;  A.pe = pe;
  A.emb_in_t = cvt(emb_in, D_, D_);  A.bemb_in = bemb_in;
  A.emb_tg = emb_tg;                 A.bemb_tg = bemb_tg;

  for (int l = 0; l < 2; ++l) {
    cvt_attn(A.enc[l].attn);
    A.enc[l].ffn.w1 = cvt(nf(), D_, DFF_);  A.enc[l].ffn.b1 = nf();
    A.enc[l].ffn.w2 = cvt(nf(), DFF_, D_);  A.enc[l].ffn.b2 = nf();
    A.enc[l].ln0g = nf(); A.enc[l].ln0b = nf();
    A.enc[l].ln1g = nf(); A.enc[l].ln1b = nf();
  }
  for (int l = 0; l < 2; ++l) {
    cvt_attn(A.dec[l].attn);
    A.dec[l].ffn.w1 = cvt(nf(), D_, DFF_);  A.dec[l].ffn.b1 = nf();
    A.dec[l].ffn.w2 = cvt(nf(), DFF_, D_);  A.dec[l].ffn.b2 = nf();
    cvt_attn(A.dec[l].cross);
    A.dec[l].ln0g = nf(); A.dec[l].ln0b = nf();
    A.dec[l].ln1g = nf(); A.dec[l].ln1b = nf();
    A.dec[l].ln2g = nf(); A.dec[l].ln2b = nf();
  }
  A.wk1 = nf(); A.bk1 = nf(); A.wk2 = nf(); A.bk2 = nf();
  A.wdec = nf(); A.bdec = nf();
  A.out = (float*)d_out;

  // preprocessing
  k_pe<<<(STEPS_ * D_ + 255) / 256, 256, 0, stream>>>(pe);
  k_fbehavior<<<(B_ * STEPS_ * 64 + 255) / 256, 256, 0, stream>>>(f_ego, re, wc, bc, fb);
  k_xin<<<(KB_ * STEPS_ * 64 + 255) / 256, 256, 0, stream>>>(fb, z, wie, bie, x_in);

  // fused transformer + head: one workgroup per sequence
  k_transformer<<<KB_, 256, SMEM_BYTES, stream>>>(A);
}
